// SupervisedMCR2Loss_31164282699887
// MI455X (gfx1250) — matrix operand order repaired and checked by reading, not compiled, not run on hardware
//
#include <hip/hip_runtime.h>
#include <hip/hip_bf16.h>
#include <math.h>

#define M_SAMPLES 65536
#define N_DIM     512
#define K_CLS     10
#define EPS_SQ    0.5f

typedef __attribute__((ext_vector_type(16))) __bf16 v16bf;
typedef __attribute__((ext_vector_type(8)))  float  v8f;

// ---- workspace layout (float units) ----
// covs    : 10 * 512*512 f32   (fully written by cov kernel; no pre-zero needed)
// meta    : 64 ints  [0..15]=counts [16..31]=cursor [32..47]=offs [48..63]=nchunks
// logdets : 16 f32
// idxbuf  : IDX_CAP u32 (class-partitioned sample indices, 0xFFFFFFFF = pad)
// S       : 11 * 512*512 f32  (shifted matrices)
// Zh      : (optional, if ws_size allows) M*N bf16 copy of Z
#define COV_FLOATS ((size_t)K_CLS * N_DIM * N_DIM)
#define S_FLOATS   ((size_t)(K_CLS + 1) * N_DIM * N_DIM)
#define IDX_CAP    (M_SAMPLES + 16 * 32)
#define BASE_FLOATS (COV_FLOATS + 80 + (size_t)IDX_CAP + S_FLOATS)

__device__ __forceinline__ unsigned f2bf(float f) {
    unsigned u = __float_as_uint(f);
    u += 0x7FFFu + ((u >> 16) & 1u);   // round-to-nearest-even
    return u >> 16;
}

union Pack { unsigned w[8]; v16bf v; };

// ---------------- setup kernels ----------------

__global__ void init_kernel(int* __restrict__ meta, unsigned* __restrict__ idxbuf) {
    int i = blockIdx.x * blockDim.x + threadIdx.x;
    if (i < 64) meta[i] = 0;
    if (i < IDX_CAP) idxbuf[i] = 0xFFFFFFFFu;
}

__global__ void count_kernel(const int* __restrict__ y, int* __restrict__ meta) {
    int m = blockIdx.x * blockDim.x + threadIdx.x;
    if (m < M_SAMPLES) atomicAdd(&meta[y[m]], 1);
}

__global__ void scan_kernel(int* __restrict__ meta) {
    if (threadIdx.x == 0 && blockIdx.x == 0) {
        int off = 0;
        for (int k = 0; k < K_CLS; ++k) {
            const int c   = meta[k];
            const int nch = (c + 31) >> 5;
            meta[32 + k] = off;      // padded start
            meta[48 + k] = nch;      // 32-sample chunks
            off += nch << 5;
        }
    }
}

__global__ void scatter_kernel(const int* __restrict__ y, int* __restrict__ meta,
                               unsigned* __restrict__ idxbuf) {
    int m = blockIdx.x * blockDim.x + threadIdx.x;
    if (m < M_SAMPLES) {
        const int k   = y[m];
        const int pos = meta[32 + k] + atomicAdd(&meta[16 + k], 1);
        idxbuf[pos] = (unsigned)m;
    }
}

// one-time fp32 -> packed bf16 conversion (used when ws_size permits)
__global__ void convert_kernel(const float* __restrict__ Z, unsigned* __restrict__ Zp) {
    size_t i = (size_t)blockIdx.x * blockDim.x + threadIdx.x;
    if (i < (size_t)M_SAMPLES * N_DIM / 2) {
        const float2 f = ((const float2*)Z)[i];
        Zp[i] = f2bf(f.x) | (f2bf(f.y) << 16);
    }
}

// ---------------- WMMA Gram kernel ----------------
// Fragment layouts per ISA 7.12.2 (16-bit, wave32):
//  A 16x32: lane%16 = row,  K = (v>=4)*16 + (lane>=16)*8 + (v&3)*2 + h
//  B 32x16: lane%16 = col,  K = (v>=4)*16 + (lane>=16)*8 + (v&3)*2 + h

template <bool PRE, bool CHECK>
__device__ __forceinline__ void load_frags(const float* __restrict__ Z,
                                           const unsigned short* __restrict__ Zh,
                                           const unsigned* __restrict__ idxbuf,
                                           int base, int hi, int gi, int gj,
                                           unsigned* aw, unsigned* bw) {
    alignas(16) unsigned ia[16], ib[16];
    // chunk index lists via B128 loads (base is 128B aligned)
    *(uint4*)(ia + 0)  = *(const uint4*)(idxbuf + base + hi * 8);
    *(uint4*)(ia + 4)  = *(const uint4*)(idxbuf + base + hi * 8 + 4);
    *(uint4*)(ia + 8)  = *(const uint4*)(idxbuf + base + 16 + hi * 8);
    *(uint4*)(ia + 12) = *(const uint4*)(idxbuf + base + 16 + hi * 8 + 4);
    *(uint4*)(ib + 0)  = *(const uint4*)(idxbuf + base + hi * 8);
    *(uint4*)(ib + 4)  = *(const uint4*)(idxbuf + base + hi * 8 + 4);
    *(uint4*)(ib + 8)  = *(const uint4*)(idxbuf + base + 16 + hi * 8);
    *(uint4*)(ib + 12) = *(const uint4*)(idxbuf + base + 16 + hi * 8 + 4);
    #pragma unroll
    for (int v = 0; v < 8; ++v) {
        const int ca = ((v & 4) ? 8 : 0) + (v & 3) * 2;
        const int cb = v * 2;
        const unsigned ra0 = ia[ca], ra1 = ia[ca + 1];
        const unsigned rb0 = ib[cb], rb1 = ib[cb + 1];
        if (PRE) {
            unsigned a0, a1, b0, b1;
            if (CHECK) {
                a0 = (ra0 < M_SAMPLES) ? (unsigned)Zh[(size_t)ra0 * N_DIM + gi] : 0u;
                a1 = (ra1 < M_SAMPLES) ? (unsigned)Zh[(size_t)ra1 * N_DIM + gi] : 0u;
                b0 = (rb0 < M_SAMPLES) ? (unsigned)Zh[(size_t)rb0 * N_DIM + gj] : 0u;
                b1 = (rb1 < M_SAMPLES) ? (unsigned)Zh[(size_t)rb1 * N_DIM + gj] : 0u;
            } else {
                a0 = Zh[(size_t)ra0 * N_DIM + gi];
                a1 = Zh[(size_t)ra1 * N_DIM + gi];
                b0 = Zh[(size_t)rb0 * N_DIM + gj];
                b1 = Zh[(size_t)rb1 * N_DIM + gj];
            }
            aw[v] = a0 | (a1 << 16);
            bw[v] = b0 | (b1 << 16);
        } else {
            float a0, a1, b0, b1;
            if (CHECK) {
                a0 = (ra0 < M_SAMPLES) ? Z[(size_t)ra0 * N_DIM + gi] : 0.0f;
                a1 = (ra1 < M_SAMPLES) ? Z[(size_t)ra1 * N_DIM + gi] : 0.0f;
                b0 = (rb0 < M_SAMPLES) ? Z[(size_t)rb0 * N_DIM + gj] : 0.0f;
                b1 = (rb1 < M_SAMPLES) ? Z[(size_t)rb1 * N_DIM + gj] : 0.0f;
            } else {
                a0 = Z[(size_t)ra0 * N_DIM + gi];
                a1 = Z[(size_t)ra1 * N_DIM + gi];
                b0 = Z[(size_t)rb0 * N_DIM + gj];
                b1 = Z[(size_t)rb1 * N_DIM + gj];
            }
            aw[v] = f2bf(a0) | (f2bf(a1) << 16);
            bw[v] = f2bf(b0) | (f2bf(b1) << 16);
        }
    }
}

template <bool PRE>
__device__ __forceinline__ v8f do_chunk(const float* __restrict__ Z,
                                        const unsigned short* __restrict__ Zh,
                                        const unsigned* __restrict__ idxbuf,
                                        int base, bool full, int hi, int gi, int gj, v8f acc) {
    Pack a, b;
    if (full) load_frags<PRE, false>(Z, Zh, idxbuf, base, hi, gi, gj, a.w, b.w);
    else      load_frags<PRE, true >(Z, Zh, idxbuf, base, hi, gi, gj, a.w, b.w);
    return __builtin_amdgcn_wmma_f32_16x16x32_bf16(false, a.v, false, b.v,
                                                   (short)0, acc, false, false);
}

// 528 workgroups: lower-triangular 16x16 tiles of the 512x512 Gram matrices.
// 8 waves each, classes processed sequentially (single accumulator pair per wave).
template <bool PRE>
__global__ __launch_bounds__(256) void cov_wmma_kernel(const float* __restrict__ Z,
                                                       const unsigned short* __restrict__ Zh,
                                                       const int* __restrict__ meta,
                                                       const unsigned* __restrict__ idxbuf,
                                                       float* __restrict__ covs) {
    __shared__ float red[256];
    const int tid  = threadIdx.x;
    const int lane = tid & 31;
    const int wave = tid >> 5;
    const int row  = lane & 15;
    const int hi   = lane >> 4;

    // triangular tile decode: b -> (ti >= tj)
    const int b = (int)blockIdx.x;
    int ti = (int)((sqrtf(8.0f * (float)b + 1.0f) - 1.0f) * 0.5f);
    while ((ti + 1) * (ti + 2) / 2 <= b) ++ti;
    while (ti * (ti + 1) / 2 > b) --ti;
    const int tj = b - ti * (ti + 1) / 2;
    const int i0 = ti * 16, j0 = tj * 16;
    const int gi = i0 + row, gj = j0 + row;

    const v8f vzero = {0.f,0.f,0.f,0.f,0.f,0.f,0.f,0.f};

    for (int k = 0; k < K_CLS; ++k) {
        const int cnt  = meta[k];
        const int off0 = meta[32 + k];
        const int nch  = meta[48 + k];
        v8f acc0 = vzero, acc1 = vzero;
        int ch = wave;
        for (; ch + 8 < nch; ch += 16) {
            acc0 = do_chunk<PRE>(Z, Zh, idxbuf, off0 + ch * 32,
                                 (ch * 32 + 32) <= cnt, hi, gi, gj, acc0);
            acc1 = do_chunk<PRE>(Z, Zh, idxbuf, off0 + (ch + 8) * 32,
                                 ((ch + 8) * 32 + 32) <= cnt, hi, gi, gj, acc1);
        }
        if (ch < nch)
            acc0 = do_chunk<PRE>(Z, Zh, idxbuf, off0 + ch * 32,
                                 (ch * 32 + 32) <= cnt, hi, gi, gj, acc0);
        const v8f acc = acc0 + acc1;

        // cross-wave reduction. C layout: VGPR v -> row v + hi*8, lane%16 -> col.
        red[tid] = 0.0f;
        __syncthreads();
        #pragma unroll
        for (int v = 0; v < 8; ++v)
            atomicAdd(&red[(v + hi * 8) * 16 + row], acc[v]);
        __syncthreads();
        {
            const int r = tid >> 4, c = tid & 15;
            const float val = red[tid];
            float* ck = covs + (size_t)k * N_DIM * N_DIM;
            ck[(size_t)(i0 + r) * N_DIM + (j0 + c)] = val;
            if (ti != tj)
                ck[(size_t)(j0 + c) * N_DIM + (i0 + r)] = val;   // symmetric mirror
        }
        __syncthreads();
    }
}

// ---------------- logdet stages ----------------

__global__ void build_shifted_kernel(const float* __restrict__ covs,
                                     const int* __restrict__ meta,
                                     float* __restrict__ S) {
    const size_t per = (size_t)N_DIM * N_DIM;
    size_t idx = (size_t)blockIdx.x * blockDim.x + threadIdx.x;
    if (idx >= (size_t)(K_CLS + 1) * per) return;
    const int    mat = (int)(idx / per);
    const size_t e   = idx % per;
    const int r = (int)(e / N_DIM), c = (int)(e % N_DIM);
    float val;
    if (mat < K_CLS) {
        const float cnt   = fmaxf((float)meta[mat], 1.0f);
        const float alpha = (float)N_DIM / (cnt * EPS_SQ);
        val = covs[(size_t)mat * per + e] * alpha;
    } else {
        float s = 0.0f;
        #pragma unroll
        for (int k = 0; k < K_CLS; ++k) s += covs[(size_t)k * per + e];
        const float alpha = (float)N_DIM / ((float)M_SAMPLES * EPS_SQ);
        val = s * alpha;
    }
    if (r == c) val += 1.0f;
    S[idx] = val;
}

__global__ __launch_bounds__(256) void chol_logdet_kernel(float* __restrict__ S,
                                                          float* __restrict__ logdets) {
    float* A = S + (size_t)blockIdx.x * N_DIM * N_DIM;
    __shared__ float sh_ljj;
    __shared__ float sh_ld;
    if (threadIdx.x == 0) sh_ld = 0.0f;
    __syncthreads();
    for (int j = 0; j < N_DIM; ++j) {
        if (threadIdx.x == 0) {
            const float d = A[(size_t)j * N_DIM + j];
            const float l = sqrtf(fmaxf(d, 1e-30f));
            sh_ljj = l;
            sh_ld += 2.0f * logf(l);
        }
        __syncthreads();
        const float inv = 1.0f / sh_ljj;
        for (int i = j + 1 + threadIdx.x; i < N_DIM; i += 256)
            A[(size_t)i * N_DIM + j] *= inv;
        __syncthreads();
        for (int i = j + 1 + threadIdx.x; i < N_DIM; i += 256) {
            const float lij = A[(size_t)i * N_DIM + j];
            for (int c = j + 1; c <= i; ++c)
                A[(size_t)i * N_DIM + c] -= lij * A[(size_t)c * N_DIM + j];
        }
        __syncthreads();
    }
    if (threadIdx.x == 0) logdets[blockIdx.x] = sh_ld;
}

__global__ void finalize_kernel(const float* __restrict__ logdets,
                                const int* __restrict__ meta,
                                float* __restrict__ out) {
    if (threadIdx.x == 0 && blockIdx.x == 0) {
        const float R = 0.5f * logdets[K_CLS];
        float Rc = 0.0f;
        #pragma unroll
        for (int k = 0; k < K_CLS; ++k) {
            const float gamma = (float)meta[k] / (float)M_SAMPLES;
            Rc += gamma * 0.5f * logdets[k];
        }
        out[0] = -(R - Rc);
    }
}

// ---------------- launcher ----------------

extern "C" void kernel_launch(void* const* d_in, const int* in_sizes, int n_in,
                              void* d_out, int out_size, void* d_ws, size_t ws_size,
                              hipStream_t stream) {
    const float* Z = (const float*)d_in[0];
    const int*   y = (const int*)d_in[1];
    float* ws      = (float*)d_ws;

    float*    covs    = ws;
    int*      meta    = (int*)(ws + COV_FLOATS);
    float*    logdets = ws + COV_FLOATS + 64;
    unsigned* idxbuf  = (unsigned*)(ws + COV_FLOATS + 80);
    float*    S       = ws + COV_FLOATS + 80 + IDX_CAP;
    unsigned short* Zh = (unsigned short*)(ws + BASE_FLOATS);
    float*    out     = (float*)d_out;

    const size_t need_pre = BASE_FLOATS * 4 + (size_t)M_SAMPLES * N_DIM * 2;
    const bool use_pre = (ws_size >= need_pre);

    init_kernel<<<(IDX_CAP + 255) / 256, 256, 0, stream>>>(meta, idxbuf);
    count_kernel<<<M_SAMPLES / 256, 256, 0, stream>>>(y, meta);
    scan_kernel<<<1, 32, 0, stream>>>(meta);
    scatter_kernel<<<M_SAMPLES / 256, 256, 0, stream>>>(y, meta, idxbuf);

    const unsigned ntiles = (N_DIM / 16) * (N_DIM / 16 + 1) / 2;   // 528
    if (use_pre) {
        const size_t pairs = (size_t)M_SAMPLES * N_DIM / 2;
        convert_kernel<<<(unsigned)((pairs + 255) / 256), 256, 0, stream>>>(Z, (unsigned*)Zh);
        cov_wmma_kernel<true><<<ntiles, 256, 0, stream>>>(Z, Zh, meta, idxbuf, covs);
    } else {
        cov_wmma_kernel<false><<<ntiles, 256, 0, stream>>>(Z, Zh, meta, idxbuf, covs);
    }

    const size_t tot = (size_t)(K_CLS + 1) * N_DIM * N_DIM;
    build_shifted_kernel<<<(unsigned)((tot + 255) / 256), 256, 0, stream>>>(covs, meta, S);
    chol_logdet_kernel<<<K_CLS + 1, 256, 0, stream>>>(S, logdets);
    finalize_kernel<<<1, 64, 0, stream>>>(logdets, meta, out);
}
